// S4Layer_3375844295113
// MI455X (gfx1250) — matrix operand order repaired and checked by reading, not compiled, not run on hardware
//
#include <hip/hip_runtime.h>

// S4 layer for MI455X (gfx1250): K-gen -> Toeplitz conv via bf16 WMMA -> GLU proj via bf16 WMMA.
// B=16 matches WMMA N=16; entire working set (x 67MB, y 32MB, W 2MB, K 2MB) fits the 192MB L2.

typedef __attribute__((ext_vector_type(16))) __bf16 v16bf;
typedef __attribute__((ext_vector_type(8)))  __bf16 v8bf;
typedef __attribute__((ext_vector_type(8)))  float  v8f;

#define B_SZ 16
#define LEN  2048
#define HDIM 512
#define NST  64
#define USTRIDE 2056   // 2048 + 8 pad: keeps 16B alignment, staggers LDS banks

__device__ __forceinline__ float fast_sigmoid(float x) {
    // branch-free: v_exp_f32 + v_rcp_f32 (no libm slow path)
    return __builtin_amdgcn_rcpf(1.0f + __expf(-x));
}
__device__ __forceinline__ float gelu_tanh(float x) {
    // 0.5x(1+tanh(z)) == x * sigmoid(2z),  z = 0.79788456(x + 0.044715 x^3)
    float z2 = 1.5957691216057308f * (x + 0.044715f * x * x * x);
    return x * fast_sigmoid(z2);
}

__device__ __forceinline__ v16bf load_pair16(const __bf16* p, int gap) {
    v8bf lo = *(const v8bf*)(p);
    v8bf hi = *(const v8bf*)(p + gap);
    return __builtin_shufflevector(lo, hi,
             0,1,2,3,4,5,6,7,8,9,10,11,12,13,14,15);
}
__device__ __forceinline__ v16bf load_k_run(const __bf16* kp) {
    v16bf bm;
#pragma unroll
    for (int i = 0; i < 16; ++i) bm[i] = kp[i];
    return bm;
}

// ---------------- Pass A: SSM kernel K[h][l] (bf16) ----------------
__global__ void __launch_bounds__(256) s4_kernelgen(
    const float* __restrict__ log_dt, const float* __restrict__ log_A_real,
    const float* __restrict__ A_imag, const float* __restrict__ B_re,
    const float* __restrict__ B_im,  const float* __restrict__ C_re,
    const float* __restrict__ C_im,  __bf16* __restrict__ Kg)
{
    __shared__ float s_wr[NST], s_wi[NST], s_ar[NST], s_ai[NST];
    const int h = blockIdx.x;
    const int t = threadIdx.x;
    if (t < NST) {
        const int n = t;
        float dt  = __expf(log_dt[h]);
        float Are = -__expf(log_A_real[h * NST + n]);
        float Aim = A_imag[h * NST + n];
        float dAr = dt * Are, dAi = dt * Aim;
        float er  = __expf(dAr);
        float sn, cs; __sincosf(dAi, &sn, &cs);
        float numr = er * cs - 1.0f, numi = er * sn;      // exp(dtA)-1
        float inv  = 1.0f / (Are * Are + Aim * Aim);
        float qr = (numr * Are + numi * Aim) * inv;       // /A
        float qi = (numi * Are - numr * Aim) * inv;
        float br = B_re[h * NST + n], bi = B_im[h * NST + n];
        float dBr = qr * br - qi * bi, dBi = qr * bi + qi * br;
        float cr = C_re[h * NST + n], ci = C_im[h * NST + n];
        s_wr[n] = cr * dBr - ci * dBi;                    // w = Cc*dB
        s_wi[n] = cr * dBi + ci * dBr;
        s_ar[n] = dAr; s_ai[n] = dAi;
    }
    __syncthreads();
    for (int l0 = 0; l0 < LEN; l0 += 256) {
        const int l = l0 + t;
        const float fl = (float)l;
        float acc = 0.0f;
#pragma unroll 4
        for (int n = 0; n < NST; ++n) {
            float er = __expf(s_ar[n] * fl);
            float sn, cs; __sincosf(s_ai[n] * fl, &sn, &cs);
            acc += er * (s_wr[n] * cs - s_wi[n] * sn);
        }
        Kg[h * LEN + l] = (__bf16)(2.0f * acc);
    }
}

// ---------------- Pass A2: one-shot W fp32 -> bf16 (reused by 2048 col tiles) ----
__global__ void __launch_bounds__(256) s4_wconv(
    const float* __restrict__ W, __bf16* __restrict__ Wb)
{
    int idx = blockIdx.x * 256 + threadIdx.x;             // 2H*H = 524288 elements
    Wb[idx] = (__bf16)W[idx];
}

// ---------------- Pass B: causal Toeplitz conv + D-skip + GELU ----------------
// One workgroup per head h.
//   u staged bf16 [b][l] (row stride USTRIDE) -> A operand = two aligned v8bf LDS loads.
//   K staged REVERSED + 32-zero tail          -> B operand = branch-free forward run.
//   D[b][lout] = sum_lin u[b][lin] * K[lout-lin];  y stored bf16 at Yg[c*H + h].
__global__ void __launch_bounds__(256) s4_conv(
    const float* __restrict__ x, const __bf16* __restrict__ Kg,
    const float* __restrict__ Dp, __bf16* __restrict__ Yg)
{
    __shared__ __bf16 u_lds[B_SZ * USTRIDE];   // 65,792 B
    __shared__ __bf16 krev_lds[LEN + 32];      //  4,160 B  (krev[m]=K[L-1-m]; tail=0)
    const int h = blockIdx.x;
    const int t = threadIdx.x;

    for (int idx = t; idx < B_SZ * LEN; idx += 256) {
        int b = idx >> 11, l = idx & (LEN - 1);
        const float* src = &x[((size_t)b * LEN + l) * HDIM + h];
        __builtin_prefetch(src + 256 * HDIM, 0, 1);       // global_prefetch_b8
        u_lds[b * USTRIDE + l] = (__bf16)(*src);          // L2-served gather
    }
    for (int m = t; m < LEN + 32; m += 256)
        krev_lds[m] = (m < LEN) ? Kg[(size_t)h * LEN + (LEN - 1 - m)] : (__bf16)0.0f;
    __syncthreads();

    const float Dh = Dp[h];
    const int wave  = __builtin_amdgcn_readfirstlane(t >> 5);  // wave-uniform
    const int lane  = t & 31;
    const int laneM = lane & 15, hiHalf = lane >> 4;
    const __bf16* urow = u_lds + laneM * USTRIDE;         // A row: b = laneM
    const int aoff = hiHalf << 4;                          // A k-offset per half

    for (int ti = wave; ti < LEN / 16; ti += 8) {         // 16-col output tiles
        v8f acc0 = {}, acc1 = {};
        const int rowbase = 16 * ti;
        // B operand base: K[lout - lin] = krev[(L-1) - (rowbase+laneM) + lin]
        const __bf16* kb0 = krev_lds + (LEN - 1 - rowbase - laneM) + (hiHalf << 4);
        const int nchunks = ti / 2 + 1;                   // lower-triangular k extent
        int kc = 0;
        for (; kc + 2 <= nchunks; kc += 2) {              // 2 indep accumulator chains
            int lin0 = kc << 5, lin1 = lin0 + 32;
            acc0 = __builtin_amdgcn_wmma_f32_16x16x32_bf16(
                       false, load_pair16(urow + lin0 + aoff, 8), false,
                       load_k_run(kb0 + lin0), (short)0, acc0, false, false);
            acc1 = __builtin_amdgcn_wmma_f32_16x16x32_bf16(
                       false, load_pair16(urow + lin1 + aoff, 8), false,
                       load_k_run(kb0 + lin1), (short)0, acc1, false, false);
        }
        if (kc < nchunks) {                               // uniform scalar tail
            int lin0 = kc << 5;
            acc0 = __builtin_amdgcn_wmma_f32_16x16x32_bf16(
                       false, load_pair16(urow + lin0 + aoff, 8), false,
                       load_k_run(kb0 + lin0), (short)0, acc0, false, false);
        }
        v8f acc = acc0 + acc1;
#pragma unroll
        for (int r = 0; r < 8; ++r) {                     // C/D: M=b=r+8*hiHalf, N=lout
            int b    = r + (hiHalf << 3);
            int lout = rowbase + laneM;
            float u = (float)u_lds[b * USTRIDE + lout];
            float y = gelu_tanh(acc[r] + Dh * u);
            Yg[((size_t)b * LEN + lout) * HDIM + h] = (__bf16)y;
        }
    }
}

// ---------------- Pass C: z = W*y + b, GLU, write (B,L,H) fp32 ----------------
// Each wave: one 16-col tile c; computes the GLU "a" row tile (o) AND matching
// "g" row tile (o+512) sharing one B tile -> fused GLU epilogue.
// All operands bf16 -> 6x 16B loads + 2x WMMA per k-chunk.
__global__ void __launch_bounds__(256) s4_out(
    const __bf16* __restrict__ Yg, const __bf16* __restrict__ Wb,
    const float* __restrict__ bias, float* __restrict__ Out)
{
    const int t = threadIdx.x;
    const int wave = __builtin_amdgcn_readfirstlane(t >> 5);
    const int lane = t & 31;
    const int laneM = lane & 15, hiHalf = lane >> 4;
    const int to = blockIdx.x >> 8;                       // 0..31 (512/16)
    const int tc = ((blockIdx.x & 255) << 3) + wave;      // 0..2047 (B*L/16)
    const int o0 = 16 * to;
    const int c  = 16 * tc + laneM;                       // column = b*LEN + l
    const int oA = o0 + laneM;                            // A-row for this lane

    const __bf16* yc = Yg + (size_t)c * HDIM + (hiHalf << 3);
    const __bf16* wa = Wb + (size_t)oA * HDIM + (hiHalf << 3);
    const __bf16* wg = Wb + ((size_t)oA + HDIM) * HDIM + (hiHalf << 3);

    v8f accA = {}, accG = {};
    for (int kc = 0; kc < HDIM / 32; ++kc) {
        const int k0 = kc * 32;
        v16bf bm = load_pair16(yc + k0, 16);              // B: two 8-runs, table k order
        v16bf aa = load_pair16(wa + k0, 16);              // A ("a" half)
        v16bf ag = load_pair16(wg + k0, 16);              // A ("g" half)
        accA = __builtin_amdgcn_wmma_f32_16x16x32_bf16(
                   false, aa, false, bm, (short)0, accA, false, false);
        accG = __builtin_amdgcn_wmma_f32_16x16x32_bf16(
                   false, ag, false, bm, (short)0, accG, false, false);
    }
#pragma unroll
    for (int r = 0; r < 8; ++r) {
        int o = o0 + r + (hiHalf << 3);
        float za = accA[r] + bias[o];
        float zg = accG[r] + bias[o + HDIM];
        Out[(size_t)c * HDIM + o] = za * fast_sigmoid(zg);
    }
}

extern "C" void kernel_launch(void* const* d_in, const int* in_sizes, int n_in,
                              void* d_out, int out_size, void* d_ws, size_t ws_size,
                              hipStream_t stream)
{
    (void)in_sizes; (void)n_in; (void)out_size; (void)ws_size;
    const float* x          = (const float*)d_in[0];
    const float* log_dt     = (const float*)d_in[1];
    const float* log_A_real = (const float*)d_in[2];
    const float* A_imag     = (const float*)d_in[3];
    const float* B_re       = (const float*)d_in[4];
    const float* B_im       = (const float*)d_in[5];
    const float* C_re       = (const float*)d_in[6];
    const float* C_im       = (const float*)d_in[7];
    const float* Dp         = (const float*)d_in[8];
    const float* W          = (const float*)d_in[9];
    const float* bias       = (const float*)d_in[10];
    float* out = (float*)d_out;

    char* ws = (char*)d_ws;
    __bf16* Kg = (__bf16*)ws;                                   // H*L bf16   =  2 MB
    __bf16* Yg = (__bf16*)(ws + (size_t)HDIM * LEN * 2);        // B*L*H bf16 = 32 MB
    __bf16* Wb = (__bf16*)(ws + (size_t)HDIM * LEN * 2
                              + (size_t)B_SZ * LEN * HDIM * 2); // 2H*H bf16  =  1 MB

    s4_wconv<<<(2 * HDIM * HDIM) / 256, 256, 0, stream>>>(W, Wb);
    s4_kernelgen<<<HDIM, 256, 0, stream>>>(log_dt, log_A_real, A_imag,
                                           B_re, B_im, C_re, C_im, Kg);
    s4_conv<<<HDIM, 256, 0, stream>>>(x, Kg, Dp, Yg);
    s4_out<<<32 * 256, 256, 0, stream>>>(Yg, Wb, bias, out);
}
